// MultiHeadAttention_34316788695519
// MI455X (gfx1250) — matrix operand order repaired and checked by reading, not compiled, not run on hardware
//
#include <hip/hip_runtime.h>

// Problem constants (from reference)
#define B_   2
#define S_   2048
#define DM   1024
#define NH   16
#define DH   64
#define MTOT (B_ * S_)   // 4096 rows

typedef __attribute__((ext_vector_type(16))) __bf16 v16bf;
typedef __attribute__((ext_vector_type(8)))  float  v8f;
typedef __attribute__((ext_vector_type(4)))  unsigned int v4u;
typedef __attribute__((ext_vector_type(8)))  unsigned int v8u;

union FragB { v16bf v; v4u u[2]; };

__device__ __forceinline__ unsigned short f2bf(float f) {
  unsigned int u = __float_as_uint(f);
  u += 0x7FFFu + ((u >> 16) & 1u);   // round-to-nearest-even
  return (unsigned short)(u >> 16);
}

// CDNA5 async memory->LDS copy (ASYNCcnt-tracked), 16 bytes per lane.
// LDS target address = low 32 bits of the generic pointer (LDS aperture offset).
__device__ __forceinline__ void async_b128(const unsigned short* g, unsigned short* l) {
  unsigned loff = (unsigned)(unsigned long long)(uintptr_t)l;
  unsigned long long ga = (unsigned long long)(uintptr_t)g;
  asm volatile("global_load_async_to_lds_b128 %0, %1, off"
               :: "v"(loff), "v"(ga) : "memory");
}
__device__ __forceinline__ void wait_async0() {
  asm volatile("s_wait_asynccnt 0x0" ::: "memory");
}
__device__ __forceinline__ void wait_tensor0() {
#if __has_builtin(__builtin_amdgcn_s_wait_tensorcnt)
  __builtin_amdgcn_s_wait_tensorcnt(0);
#else
  asm volatile("s_wait_tensorcnt 0x0" ::: "memory");
#endif
}

// CDNA5 Tensor Data Mover: DMA a 2D tile (bf16 elements) from global memory
// into LDS.  Descriptor bit layout per CDNA5 ISA.
//   group0: [1:0]count=1 | [63:32]lds_addr | [120:64]global_addr | [127:126]type=2
//   group1: data_size=1(2B), tensor_dim0/1, tile_dim0/1, tensor_dim0_stride
__device__ __forceinline__ void tdm_load_2d(const unsigned short* gaddr,
                                            unsigned lds_off,
                                            unsigned tensor_d0, unsigned tensor_d1,
                                            unsigned tile_d0, unsigned tile_d1,
                                            unsigned stride0_elems) {
  unsigned long long ga = (unsigned long long)(uintptr_t)gaddr;
  v4u g0;
  g0[0] = 1u;                                                  // count=1 (user D#)
  g0[1] = lds_off;                                             // LDS byte address
  g0[2] = (unsigned)ga;                                        // global_addr[31:0]
  g0[3] = ((unsigned)(ga >> 32) & 0x01FFFFFFu) | 0x80000000u;  // [56:32] | type=2
  v8u g1;
  g1[0] = 0x00010000u;                                // wg_mask=0, data_size=1 (2B)
  g1[1] = (tensor_d0 & 0xFFFFu) << 16;                // tensor_dim0[15:0] @ [63:48]
  g1[2] = (tensor_d0 >> 16) | ((tensor_d1 & 0xFFFFu) << 16);
  g1[3] = (tensor_d1 >> 16) | (tile_d0 << 16);        // tile_dim0 @ [127:112]
  g1[4] = tile_d1;                                    // tile_dim1; tile_dim2=0
  g1[5] = stride0_elems;                              // tensor_dim0_stride[31:0]
  g1[6] = 0;                                          // stride0[47:32]|stride1[15:0]
  g1[7] = 0;                                          // stride1[47:16]
  asm volatile("tensor_load_to_lds %0, %1" :: "s"(g0), "s"(g1) : "memory");
}

// ---------------------------------------------------------------------------
// f32 -> bf16 elementwise convert
// ---------------------------------------------------------------------------
__global__ void cvt_f32_bf16(const float* __restrict__ src,
                             unsigned short* __restrict__ dst, int n) {
  int i = blockIdx.x * blockDim.x + threadIdx.x;
  int stride = gridDim.x * blockDim.x;
  for (; i < n; i += stride) dst[i] = f2bf(src[i]);
}

// f32 [K][N] -> bf16 [N][K] (transposed weights; K=N=1024)
__global__ void transpose_w(const float* __restrict__ W,
                            unsigned short* __restrict__ Wt) {
  int o = blockIdx.x * 256 + threadIdx.x;   // over N*K = 1M
  int n = o >> 10;
  int k = o & 1023;
  Wt[o] = f2bf(W[k * DM + n]);
}

// ---------------------------------------------------------------------------
// bf16 WMMA GEMM: C[M=4096][N=1024] = A[M][K=1024] * Bt[N][K]^T + bias
// Double-buffered LDS tiles filled by the Tensor Data Mover (TENSORcnt):
// each of the 8 waves DMAs its own 16x32 slice of the A and Bt tiles.
// mode 0: bf16 head-split  out[((b*16+h)*2048+s)*64+d]   (Q, K)
// mode 2: bf16 head-split transposed out[((b*16+h)*64+d)*2048+s] (V)
// mode 1: f32 resid[m*1024+n] = acc + bias[n] + qres[m*1024+n]
// ---------------------------------------------------------------------------
__global__ __launch_bounds__(256) void gemm_wmma(
    const unsigned short* __restrict__ A, const unsigned short* __restrict__ Bt,
    const float* __restrict__ bias, const float* __restrict__ qres,
    unsigned short* __restrict__ outb, float* __restrict__ outf, int mode) {
  __shared__ unsigned short As[2][128 * 32];
  __shared__ unsigned short Bs[2][128 * 32];
  const int K = DM;

  int m0 = blockIdx.y * 128, n0 = blockIdx.x * 128;
  int tid = threadIdx.x;
  int wid = tid >> 5, lane = tid & 31;
  int ln = lane & 15, hi = lane >> 4;
  int wm = wid & 3, wn = wid >> 2;   // 4 waves in M x 2 waves in N

  v8f zero = {};
  v8f acc[2][4];
#pragma unroll
  for (int i = 0; i < 2; i++)
#pragma unroll
    for (int j = 0; j < 4; j++) acc[i][j] = zero;

  // TDM staging: wave w owns rows [w*16, w*16+16) of both tiles
  const unsigned short* gA = A + (size_t)(m0 + wid * 16) * K;   // + k0
  const unsigned short* gB = Bt + (size_t)(n0 + wid * 16) * K;  // + k0
  unsigned aBase = (unsigned)(unsigned long long)(uintptr_t)&As[0][0];
  unsigned bBase = (unsigned)(unsigned long long)(uintptr_t)&Bs[0][0];
  unsigned ldsA = aBase + wid * 16 * 32 * 2;   // + buf*8192
  unsigned ldsB = bBase + wid * 16 * 32 * 2;

  // prologue: DMA k-tile 0 into buffer 0
  tdm_load_2d(gA, ldsA, K, MTOT, 32, 16, K);
  tdm_load_2d(gB, ldsB, K, DM, 32, 16, K);
  wait_tensor0();
  __syncthreads();

  int cur = 0;
  for (int k0 = 0; k0 < K; k0 += 32) {
    int nxt = cur ^ 1;
    if (k0 + 32 < K) {   // DMA next k-tile into the other buffer
      tdm_load_2d(gA + k0 + 32, ldsA + nxt * 8192, K, MTOT, 32, 16, K);
      tdm_load_2d(gB + k0 + 32, ldsB + nxt * 8192, K, DM, 32, 16, K);
      __builtin_prefetch(gA + k0 + 64, 0, 3);   // warm caches for tile after next
      __builtin_prefetch(gB + k0 + 64, 0, 3);
    }

    FragB af[2], bfr[4];
#pragma unroll
    for (int mt = 0; mt < 2; mt++) {
      const unsigned short* p = &As[cur][(wm * 32 + mt * 16 + ln) * 32];
      af[mt].u[0] = *(const v4u*)(p + hi * 8);        // k = hi*8 .. +7
      af[mt].u[1] = *(const v4u*)(p + 16 + hi * 8);   // k = 16+hi*8 .. +7
    }
#pragma unroll
    for (int nt = 0; nt < 4; nt++) {
      const unsigned short* p = &Bs[cur][(wn * 64 + nt * 16 + ln) * 32 + hi * 8];
      bfr[nt].u[0] = *(const v4u*)(p);                // k = hi*8 .. +7
      bfr[nt].u[1] = *(const v4u*)(p + 16);           // k = 16+hi*8 .. +7
    }
#pragma unroll
    for (int mt = 0; mt < 2; mt++)
#pragma unroll
      for (int nt = 0; nt < 4; nt++)
        acc[mt][nt] = __builtin_amdgcn_wmma_f32_16x16x32_bf16(
            false, af[mt].v, false, bfr[nt].v, (short)0, acc[mt][nt], false, false);

    wait_tensor0();    // this wave's DMAs into buffer nxt have landed
    __syncthreads();   // everyone done reading cur + all slices visible
    cur = nxt;
  }

  // epilogue: C layout = VGPR r: rows r (lanes 0-15) / r+8 (lanes 16-31)
#pragma unroll
  for (int mt = 0; mt < 2; mt++) {
#pragma unroll
    for (int nt = 0; nt < 4; nt++) {
#pragma unroll
      for (int r = 0; r < 8; r++) {
        int m = m0 + wm * 32 + mt * 16 + r + hi * 8;
        int n = n0 + wn * 64 + nt * 16 + ln;
        float val = acc[mt][nt][r] + bias[n];
        if (mode == 1) {
          outf[(size_t)m * DM + n] = val + qres[(size_t)m * DM + n];
        } else {
          int b = m >> 11, s = m & (S_ - 1);
          int h = n >> 6, d = n & (DH - 1);
          size_t idx = (mode == 0)
                           ? (((size_t)(b * NH + h) * S_ + s) * DH + d)
                           : (((size_t)(b * NH + h) * DH + d) * S_ + s);
          outb[idx] = f2bf(val);
        }
      }
    }
  }
}

// ---------------------------------------------------------------------------
// Flash attention: Qh [b,h,s,64] bf16, Kh [b,h,s,64] bf16, Vt [b,h,64,s] bf16
// -> attnb [b,s,h*64] bf16.  4 waves/WG, 16 queries/wave, 64-key blocks,
// double-buffered K/V tiles via GLOBAL_LOAD_ASYNC_TO_LDS (ASYNCcnt path).
// ---------------------------------------------------------------------------
__global__ __launch_bounds__(128) void attn_wmma(
    const unsigned short* __restrict__ Qh, const unsigned short* __restrict__ Kh,
    const unsigned short* __restrict__ Vt, unsigned short* __restrict__ attnb) {
  __shared__ unsigned short Ks[2][64][64];   // [buf][key][d]
  __shared__ unsigned short Vs[2][64][64];   // [buf][d][key]
  __shared__ unsigned short Ps[4][16][64];   // per-wave P tile [q][key]

  int bh = blockIdx.y;     // b*16 + h
  int qblk = blockIdx.x;   // 0..31
  int tid = threadIdx.x;
  int wid = tid >> 5, lane = tid & 31;
  int ln = lane & 15, hi = lane >> 4;
  int q0 = qblk * 64 + wid * 16;

  const size_t bhBase = (size_t)bh * S_ * DH;   // 131072 elems

  // Q A-fragments, resident for the whole kernel (d chunks 0-31, 32-63)
  FragB qf[2];
#pragma unroll
  for (int c = 0; c < 2; c++) {
    const unsigned short* p = Qh + bhBase + (size_t)(q0 + ln) * DH + c * 32 + hi * 8;
    qf[c].u[0] = *(const v4u*)p;
    qf[c].u[1] = *(const v4u*)(p + 16);
  }

  v8f zero = {};
  v8f accO[4];
#pragma unroll
  for (int dt = 0; dt < 4; dt++) accO[dt] = zero;
  float mrow[8], lrow[8];
#pragma unroll
  for (int r = 0; r < 8; r++) { mrow[r] = -1e30f; lrow[r] = 0.0f; }

  int sr = tid >> 1, sh = tid & 1;   // 2 threads per 64-elem row, 64B each
  const unsigned short* gK = Kh + bhBase + (size_t)sr * DH + sh * 32;   // + kb*64
  const unsigned short* gV = Vt + bhBase + (size_t)sr * S_ + sh * 32;   // + kb
  unsigned short* lK = &Ks[0][sr][sh * 32];   // buf stride = 64*64
  unsigned short* lV = &Vs[0][sr][sh * 32];

  // prologue: stage key-block 0 into buffer 0
#pragma unroll
  for (int j = 0; j < 4; j++) {
    async_b128(gK + j * 8, lK + j * 8);
    async_b128(gV + j * 8, lV + j * 8);
  }
  wait_async0();
  __syncthreads();

  int cur = 0;
  for (int kb = 0; kb < S_; kb += 64) {
    int nxt = cur ^ 1;
    if (kb + 64 < S_) {   // stage next key-block
      const unsigned short* nK = gK + (size_t)(kb + 64) * DH;
      const unsigned short* nV = gV + (kb + 64);
#pragma unroll
      for (int j = 0; j < 4; j++) {
        async_b128(nK + j * 8, lK + nxt * (64 * 64) + j * 8);
        async_b128(nV + j * 8, lV + nxt * (64 * 64) + j * 8);
      }
      __builtin_prefetch(gK + (size_t)(kb + 128) * DH, 0, 3);
      __builtin_prefetch(gV + (kb + 128), 0, 3);
    }

    // scores = Q @ K^T : 4 N-tiles of 16 keys, 2 k-chunks of d
    v8f sc[4];
#pragma unroll
    for (int nt = 0; nt < 4; nt++) {
      sc[nt] = zero;
#pragma unroll
      for (int c = 0; c < 2; c++) {
        FragB kf;
        const unsigned short* p = &Ks[cur][nt * 16 + ln][c * 32 + hi * 8];
        kf.u[0] = *(const v4u*)p;
        kf.u[1] = *(const v4u*)(p + 16);
        sc[nt] = __builtin_amdgcn_wmma_f32_16x16x32_bf16(
            false, qf[c].v, false, kf.v, (short)0, sc[nt], false, false);
      }
    }

    // online softmax per row (row M = r + hi*8; stats reduce across 16 lanes)
#pragma unroll
    for (int r = 0; r < 8; r++) {
      float mx = -1e30f;
#pragma unroll
      for (int nt = 0; nt < 4; nt++) {
        sc[nt][r] *= 0.125f;   // 1/sqrt(64)
        mx = fmaxf(mx, sc[nt][r]);
      }
#pragma unroll
      for (int msk = 1; msk < 16; msk <<= 1)
        mx = fmaxf(mx, __shfl_xor(mx, msk, 32));
      float mnew = fmaxf(mrow[r], mx);
      float alpha = __expf(mrow[r] - mnew);
      float rs = 0.0f;
      int M = r + hi * 8;
#pragma unroll
      for (int nt = 0; nt < 4; nt++) {
        float p = __expf(sc[nt][r] - mnew);
        Ps[wid][M][nt * 16 + ln] = f2bf(p);
        rs += p;
      }
#pragma unroll
      for (int msk = 1; msk < 16; msk <<= 1) rs += __shfl_xor(rs, msk, 32);
      lrow[r] = lrow[r] * alpha + rs;
      mrow[r] = mnew;
#pragma unroll
      for (int dt = 0; dt < 4; dt++) accO[dt][r] *= alpha;
    }
    // Ps is per-wave: same-wave DS ops are in-order, compiler inserts dscnt waits

    // O += P @ V : A = P (16x32 chunks), B = Vs (d-major, contiguous keys)
    FragB pf[2];
#pragma unroll
    for (int kc = 0; kc < 2; kc++) {
      const unsigned short* p = &Ps[wid][ln][kc * 32 + hi * 8];
      pf[kc].u[0] = *(const v4u*)p;
      pf[kc].u[1] = *(const v4u*)(p + 16);
    }
#pragma unroll
    for (int dt = 0; dt < 4; dt++) {
#pragma unroll
      for (int kc = 0; kc < 2; kc++) {
        FragB vf;
        const unsigned short* p = &Vs[cur][dt * 16 + ln][kc * 32 + hi * 8];
        vf.u[0] = *(const v4u*)p;
        vf.u[1] = *(const v4u*)(p + 16);
        accO[dt] = __builtin_amdgcn_wmma_f32_16x16x32_bf16(
            false, pf[kc].v, false, vf.v, (short)0, accO[dt], false, false);
      }
    }

    wait_async0();     // our async fills of buffer nxt are in LDS
    __syncthreads();   // all waves done with cur; all fills visible
    cur = nxt;
  }

  // epilogue: divide by l, write bf16 merged-head layout [b, s, h*64+d]
  int b = bh >> 4, h = bh & 15;
#pragma unroll
  for (int dt = 0; dt < 4; dt++) {
#pragma unroll
    for (int r = 0; r < 8; r++) {
      int M = r + hi * 8;
      float val = accO[dt][r] * (1.0f / lrow[r]);
      size_t idx = ((size_t)b * S_ + q0 + M) * DM + h * DH + dt * 16 + ln;
      attnb[idx] = f2bf(val);
    }
  }
}

// ---------------------------------------------------------------------------
// LayerNorm over rows of 1024 (resid f32 -> out f32)
// ---------------------------------------------------------------------------
__global__ __launch_bounds__(256) void layernorm_k(
    const float* __restrict__ resid, const float* __restrict__ gamma,
    const float* __restrict__ beta, float* __restrict__ out) {
  __shared__ float red0[8], red1[8];
  int row = blockIdx.x;
  int tid = threadIdx.x;
  const float* x = resid + (size_t)row * DM;
  float v[4];
  float s = 0.f, ss = 0.f;
#pragma unroll
  for (int j = 0; j < 4; j++) {
    v[j] = x[tid * 4 + j];
    s += v[j];
    ss += v[j] * v[j];
  }
#pragma unroll
  for (int m = 1; m < 32; m <<= 1) {
    s += __shfl_xor(s, m, 32);
    ss += __shfl_xor(ss, m, 32);
  }
  int wid = tid >> 5, lane = tid & 31;
  if (lane == 0) { red0[wid] = s; red1[wid] = ss; }
  __syncthreads();
  float ts = 0.f, tss = 0.f;
#pragma unroll
  for (int i = 0; i < 8; i++) { ts += red0[i]; tss += red1[i]; }
  float mu = ts * (1.0f / DM);
  float var = tss * (1.0f / DM) - mu * mu;
  float rs = rsqrtf(var + 1e-5f);
#pragma unroll
  for (int j = 0; j < 4; j++) {
    int c = tid * 4 + j;
    out[(size_t)row * DM + c] = (v[j] - mu) * rs * gamma[c] + beta[c];
  }
}

// ---------------------------------------------------------------------------
// Launch
// ---------------------------------------------------------------------------
extern "C" void kernel_launch(void* const* d_in, const int* in_sizes, int n_in,
                              void* d_out, int out_size, void* d_ws, size_t ws_size,
                              hipStream_t stream) {
  (void)in_sizes; (void)n_in; (void)out_size;
  const float* q     = (const float*)d_in[0];
  const float* kv    = (const float*)d_in[1];
  const float* Wq    = (const float*)d_in[2];
  const float* bq    = (const float*)d_in[3];
  const float* Wk    = (const float*)d_in[4];
  const float* bk    = (const float*)d_in[5];
  const float* Wv    = (const float*)d_in[6];
  const float* bv    = (const float*)d_in[7];
  const float* Wo    = (const float*)d_in[8];
  const float* bo    = (const float*)d_in[9];
  const float* gamma = (const float*)d_in[10];
  const float* beta  = (const float*)d_in[11];
  float* out = (float*)d_out;

  if (ws_size < 75497472u) return;   // 72 MB scratch layout below

  char* w = (char*)d_ws;
  unsigned short* qb    = (unsigned short*)(w + 0);
  unsigned short* kvb   = (unsigned short*)(w + 8388608);
  unsigned short* Wqt   = (unsigned short*)(w + 16777216);
  unsigned short* Wkt   = (unsigned short*)(w + 18874368);
  unsigned short* Wvt   = (unsigned short*)(w + 20971520);
  unsigned short* Wot   = (unsigned short*)(w + 23068672);
  unsigned short* Qh    = (unsigned short*)(w + 25165824);
  unsigned short* Kh    = (unsigned short*)(w + 33554432);
  unsigned short* Vt    = (unsigned short*)(w + 41943040);
  unsigned short* attnb = (unsigned short*)(w + 50331648);
  float*          resid = (float*)(w + 58720256);

  const int nact = MTOT * DM;   // 4,194,304
  cvt_f32_bf16<<<2048, 256, 0, stream>>>(q, qb, nact);
  cvt_f32_bf16<<<2048, 256, 0, stream>>>(kv, kvb, nact);
  transpose_w<<<4096, 256, 0, stream>>>(Wq, Wqt);
  transpose_w<<<4096, 256, 0, stream>>>(Wk, Wkt);
  transpose_w<<<4096, 256, 0, stream>>>(Wv, Wvt);
  transpose_w<<<4096, 256, 0, stream>>>(Wo, Wot);

  dim3 gg(8, 32);   // N/128 x M/128
  gemm_wmma<<<gg, 256, 0, stream>>>(qb,  Wqt, bq, nullptr, Qh, nullptr, 0);
  gemm_wmma<<<gg, 256, 0, stream>>>(kvb, Wkt, bk, nullptr, Kh, nullptr, 0);
  gemm_wmma<<<gg, 256, 0, stream>>>(kvb, Wvt, bv, nullptr, Vt, nullptr, 2);

  attn_wmma<<<dim3(32, 32), 128, 0, stream>>>(Qh, Kh, Vt, attnb);

  gemm_wmma<<<gg, 256, 0, stream>>>(attnb, Wot, bo, q, nullptr, resid, 1);
  layernorm_k<<<4096, 256, 0, stream>>>(resid, gamma, beta, out);
}
